// Cluster_assigner_24464133718152
// MI455X (gfx1250) — matrix operand order repaired and checked by reading, not compiled, not run on hardware
//
#include <hip/hip_runtime.h>
#include <hip/hip_bf16.h>

typedef _Float16 v8h  __attribute__((ext_vector_type(8)));
typedef _Float16 v16h __attribute__((ext_vector_type(16)));
typedef float    v8f  __attribute__((ext_vector_type(8)));

#define CAT16(lo, hi) __builtin_shufflevector(lo, hi, 0,1,2,3,4,5,6,7,8,9,10,11,12,13,14,15)

#define BS  64
#define SEQ 1024
#define NV  1024
#define DM  512
#define NC  32

static __device__ __forceinline__ unsigned pk2(float a, float b) {
  union { _Float16 h[2]; unsigned u; } u_;
  u_.h[0] = (_Float16)a; u_.h[1] = (_Float16)b;
  return u_.u;
}

static __device__ __forceinline__ v8f vzero8() {
  v8f z;
#pragma unroll
  for (int i = 0; i < 8; ++i) z[i] = 0.0f;
  return z;
}

static __device__ __forceinline__ v8f wmma_f16(v16h a, v16h b, v8f c) {
  return __builtin_amdgcn_wmma_f32_16x16x32_f16(false, a, false, b, (short)0, c, false, false);
}

// ---------------------------------------------------------------------------
// W (512 x 1024 f32) -> f16 copy
__global__ __launch_bounds__(256) void k_convert_w(const float* __restrict__ W,
                                                   _Float16* __restrict__ Wh) {
  int i = blockIdx.x * 256 + threadIdx.x;   // grid covers 512*1024
  Wh[i] = (_Float16)W[i];
}

// cluster_emb (32 x 512): raw f16 copy + L2-normalized f32 copy
__global__ __launch_bounds__(256) void k_prep_cluster(const float* __restrict__ ce,
                                                      _Float16* __restrict__ ch,
                                                      float* __restrict__ cn) {
  __shared__ float red[8];
  const int c = blockIdx.x, t = threadIdx.x;
  const int wave = t >> 5, lane = t & 31;
  float v0 = ce[c * DM + t];
  float v1 = ce[c * DM + 256 + t];
  float ss = v0 * v0 + v1 * v1;
#pragma unroll
  for (int k = 16; k >= 1; k >>= 1) ss += __shfl_xor(ss, k, 32);
  if (lane == 0) red[wave] = ss;
  __syncthreads();
  float tot = 0.f;
#pragma unroll
  for (int w = 0; w < 8; ++w) tot += red[w];
  const float inv = 1.0f / fmaxf(sqrtf(tot), 1e-12f);
  ch[c * DM + t]       = (_Float16)v0;
  ch[c * DM + 256 + t] = (_Float16)v1;
  cn[c * DM + t]       = v0 * inv;
  cn[c * DM + 256 + t] = v1 * inv;
}

// ---------------------------------------------------------------------------
// x_emb[b,v,d] = sum_s x[b,s,v] * W[d,s] + bias[d]
// block: 512 thr (16 waves) -> C tile 64(v) x 512(d); grid = 64 * 16
// A = x tile (v x s), staged f32->f16 transposed through double-buffered LDS
//     in 64-deep k-slabs (2 WMMA k-steps per stage, 1 barrier per slab)
// B = W_f16 (s x d column-contiguous) loaded straight from global (32B/lane)
__global__ __launch_bounds__(512) void k_embed(const float* __restrict__ x,
                                               const _Float16* __restrict__ Wh,
                                               const float* __restrict__ bias,
                                               float* __restrict__ xemb) {
  __shared__ unsigned lds_x[2][64 * 36];   // 2 x (64 v-rows, pitch 36 dw = 32 s-pairs + pad)

  const int bid = blockIdx.x;        // 0..1023
  const int b   = bid >> 4;
  const int vt  = bid & 15;
  const int v0  = vt * 64;

  const int tid  = threadIdx.x;
  const int wave = tid >> 5, lane = tid & 31;
  const int vsub = wave & 3;         // 4 x 16 v-rows
  const int d0   = (wave >> 2) * 128;// 4 x 128 d-cols
  const int lm = lane & 15, hi = lane >> 4;

  v8f acc[8];
#pragma unroll
  for (int t = 0; t < 8; ++t) acc[t] = vzero8();

  const int sp = tid >> 4;           // 0..31 -> s-pair within slab
  const int vi = (tid & 15) << 2;    // 0..60 -> 4 v rows

  auto stage = [&](int it, int buf) {
    const int s0 = it * 64;
    const float* px = x + ((size_t)b * SEQ + (s0 + 2 * sp)) * NV + (v0 + vi);
    const float4 a0 = *(const float4*)px;
    const float4 a1 = *(const float4*)(px + NV);
    unsigned* Lb = lds_x[buf];
    Lb[(vi + 0) * 36 + sp] = pk2(a0.x, a1.x);
    Lb[(vi + 1) * 36 + sp] = pk2(a0.y, a1.y);
    Lb[(vi + 2) * 36 + sp] = pk2(a0.z, a1.z);
    Lb[(vi + 3) * 36 + sp] = pk2(a0.w, a1.w);
    if (it + 1 < 16) __builtin_prefetch(px + 64 * (size_t)NV, 0, 1);
  };

  stage(0, 0);
  for (int it = 0; it < 16; ++it) {
    __syncthreads();                  // stage(it) visible; buf (it+1)&1 free
    if (it + 1 < 16) stage(it + 1, (it + 1) & 1);
    const unsigned* Lb = lds_x[it & 1];
#pragma unroll
    for (int kk = 0; kk < 2; ++kk) {
      const int s0 = it * 64 + kk * 32;
      // A fragment (16 x 32 f16) from LDS
      const int row  = vsub * 16 + lm;
      const int base = row * 36 + kk * 16 + hi * 8;
      const v8h alo = *(const v8h*)&Lb[base];
      const v8h ahi = *(const v8h*)&Lb[base + 4];
      const v16h af = CAT16(alo, ahi);
#pragma unroll
      for (int t = 0; t < 8; ++t) {
        const _Float16* wp = Wh + (size_t)(d0 + t * 16 + lm) * SEQ + s0 + hi * 16;
        const v16h bf = *(const v16h*)wp;   // 32 contiguous bytes per lane
        acc[t] = wmma_f16(af, bf, acc[t]);
      }
    }
  }

#pragma unroll
  for (int t = 0; t < 8; ++t) {
    const int d = d0 + t * 16 + lm;
    const float bv = bias[d];
#pragma unroll
    for (int r = 0; r < 8; ++r) {
      const int vrow = v0 + vsub * 16 + r + hi * 8;
      xemb[((size_t)b * NV + vrow) * DM + d] = acc[t][r] + bv;
    }
  }
}

// ---------------------------------------------------------------------------
// prob_avg (sinkhorn) + concrete mask, one block per variable v
__global__ __launch_bounds__(256) void k_prob(const float* __restrict__ xemb,
                                              const float* __restrict__ cn,
                                              const float* __restrict__ noise,
                                              float* __restrict__ out_prob,
                                              float* __restrict__ mask) {
  __shared__ float cnl[NC * DM];   // 64 KB normalized cluster matrix
  __shared__ float partial[8 * NC];
  const int v = blockIdx.x, tid = threadIdx.x;
  const int wave = tid >> 5, lane = tid & 31;

  for (int i = tid; i < NC * DM; i += 256) cnl[i] = cn[i];
  __syncthreads();

  float acc[NC];
#pragma unroll
  for (int c = 0; c < NC; ++c) acc[c] = 0.f;

  for (int bb = wave; bb < BS; bb += 8) {
    const float* rp = xemb + ((size_t)bb * NV + v) * DM;
    float rv[16];
#pragma unroll
    for (int j = 0; j < 16; ++j) rv[j] = rp[lane + 32 * j];
    float ss = 0.f;
#pragma unroll
    for (int j = 0; j < 16; ++j) ss += rv[j] * rv[j];
#pragma unroll
    for (int k = 16; k >= 1; k >>= 1) ss += __shfl_xor(ss, k, 32);
    const float inv = 1.0f / fmaxf(sqrtf(ss), 1e-12f);
#pragma unroll
    for (int c = 0; c < NC; ++c) {
      const float* cp = cnl + c * DM;
      float d = 0.f;
#pragma unroll
      for (int j = 0; j < 16; ++j) d += rv[j] * cp[lane + 32 * j];
#pragma unroll
      for (int k = 16; k >= 1; k >>= 1) d += __shfl_xor(d, k, 32);
      acc[c] += d * inv;
    }
  }

  if (lane == 0) {
#pragma unroll
    for (int c = 0; c < NC; ++c) partial[wave * NC + c] = acc[c];
  }
  __syncthreads();

  if (wave == 0) {
    float s = 0.f;
#pragma unroll
    for (int w = 0; w < 8; ++w) s += partial[w * NC + lane];
    const float pa_raw = s * (1.0f / 64.0f);
    const float q = expf(pa_raw * 20.0f);   // 1/EPSILON
    float rs = q;
#pragma unroll
    for (int k = 16; k >= 1; k >>= 1) rs += __shfl_xor(rs, k, 32);
    const float pa = q / rs;
    out_prob[v * NC + lane] = pa;

    const float nz = noise[v * NC + lane];
    const float ln = logf(nz) - log1pf(-nz);
    const float lp = logf(pa + 1e-10f) - logf(1.0f - pa + 1e-10f);
    mask[v * NC + lane] = 1.0f / (1.0f + expf(-(lp + ln) * (1.0f / 0.07f)));
  }
}

// ---------------------------------------------------------------------------
// attention for one batch per block (512 thr / 16 waves):
//   scores (WMMA) -> LDS -> softmax*mask -> A_h (f16 LDS) -> out (WMMA)
__global__ __launch_bounds__(512) void k_attn(const float* __restrict__ xemb,
                                              const _Float16* __restrict__ ch,
                                              const float* __restrict__ mask,
                                              float* __restrict__ out_part) {
  __shared__ float    sc[NC * NV];      // 128 KB score matrix
  __shared__ _Float16 Ah[NC * 1040];    // attn probs, pitch 1040 halves
  __shared__ _Float16 xT[DM * 40];      // transposed x_emb k-slab, pitch 40 halves

  const int b    = blockIdx.x;
  const int tid  = threadIdx.x;
  const int wave = tid >> 5, lane = tid & 31;
  const int lm = lane & 15, hi = lane >> 4;

  // ---- phase 1: scores[l,s] = q . x_emb * scale ----
  {
    const int l0    = (wave & 1) * 16;
    const int sbase = (wave >> 1) * 128;
    v8f acc[8];
#pragma unroll
    for (int i = 0; i < 8; ++i) acc[i] = vzero8();

    for (int kk = 0; kk < 16; ++kk) {
      const int k0 = kk * 32;
      const _Float16* cp = ch + (size_t)(l0 + lm) * DM + k0 + hi * 16;
      const v8h alo = *(const v8h*)cp;
      const v8h ahi = *(const v8h*)(cp + 8);
      const v16h af = CAT16(alo, ahi);
#pragma unroll
      for (int st = 0; st < 8; ++st) {
        const int s = sbase + st * 16 + lm;
        const float* bp = xemb + ((size_t)b * NV + s) * DM + k0 + hi * 16;
        const float4 f0 = ((const float4*)bp)[0];
        const float4 f1 = ((const float4*)bp)[1];
        const float4 f2 = ((const float4*)bp)[2];
        const float4 f3 = ((const float4*)bp)[3];
        v16h bf;
        bf[0]  = (_Float16)f0.x; bf[1]  = (_Float16)f0.y;
        bf[2]  = (_Float16)f0.z; bf[3]  = (_Float16)f0.w;
        bf[4]  = (_Float16)f1.x; bf[5]  = (_Float16)f1.y;
        bf[6]  = (_Float16)f1.z; bf[7]  = (_Float16)f1.w;
        bf[8]  = (_Float16)f2.x; bf[9]  = (_Float16)f2.y;
        bf[10] = (_Float16)f2.z; bf[11] = (_Float16)f2.w;
        bf[12] = (_Float16)f3.x; bf[13] = (_Float16)f3.y;
        bf[14] = (_Float16)f3.z; bf[15] = (_Float16)f3.w;
        acc[st] = wmma_f16(af, bf, acc[st]);
      }
    }
    const float scale = 0.04419417382415922f;   // 1/sqrt(512)
#pragma unroll
    for (int st = 0; st < 8; ++st) {
#pragma unroll
      for (int r = 0; r < 8; ++r) {
        sc[(l0 + r + hi * 8) * NV + sbase + st * 16 + lm] = acc[st][r] * scale;
      }
    }
  }
  __syncthreads();

  // ---- phase 2: per-row softmax over s, then * mask[s,l] ----
#pragma unroll
  for (int rrow = 0; rrow < 2; ++rrow) {
    const int l = wave + rrow * 16;
    float e[32];
    float mx = -3.0e38f;
#pragma unroll
    for (int j = 0; j < 32; ++j) { e[j] = sc[l * NV + lane + 32 * j]; mx = fmaxf(mx, e[j]); }
#pragma unroll
    for (int k = 16; k >= 1; k >>= 1) mx = fmaxf(mx, __shfl_xor(mx, k, 32));
    float sum = 0.f;
#pragma unroll
    for (int j = 0; j < 32; ++j) { e[j] = __expf(e[j] - mx); sum += e[j]; }
#pragma unroll
    for (int k = 16; k >= 1; k >>= 1) sum += __shfl_xor(sum, k, 32);
    const float invs = 1.0f / sum;
#pragma unroll
    for (int j = 0; j < 32; ++j) {
      const int s = lane + 32 * j;
      Ah[l * 1040 + s] = (_Float16)(e[j] * invs * mask[s * NC + l]);
    }
  }
  __syncthreads();

  // ---- phase 3: out[l,d] = sum_s A[l,s] * x_emb[b,s,d] ----
  {
    const int l0 = (wave & 1) * 16;
    const int d0 = (wave >> 1) * 64;
    v8f oc[4];
#pragma unroll
    for (int i = 0; i < 4; ++i) oc[i] = vzero8();

    for (int ks = 0; ks < 32; ++ks) {
      const int s0 = ks * 32;
      __syncthreads();
      {
        const int sp = tid >> 5;       // 0..15 -> s-pair
        const int s  = s0 + 2 * sp;
        unsigned* xtw = (unsigned*)xT;
#pragma unroll
        for (int p = 0; p < 2; ++p) {
          const int db = ((tid & 31) + p * 32) * 8;
          const float* p0 = xemb + ((size_t)b * NV + s) * DM + db;
          const float* p1 = p0 + DM;
          const float4 f0a = ((const float4*)p0)[0];
          const float4 f0b = ((const float4*)p0)[1];
          const float4 f1a = ((const float4*)p1)[0];
          const float4 f1b = ((const float4*)p1)[1];
          xtw[(db + 0) * 20 + sp] = pk2(f0a.x, f1a.x);
          xtw[(db + 1) * 20 + sp] = pk2(f0a.y, f1a.y);
          xtw[(db + 2) * 20 + sp] = pk2(f0a.z, f1a.z);
          xtw[(db + 3) * 20 + sp] = pk2(f0a.w, f1a.w);
          xtw[(db + 4) * 20 + sp] = pk2(f0b.x, f1b.x);
          xtw[(db + 5) * 20 + sp] = pk2(f0b.y, f1b.y);
          xtw[(db + 6) * 20 + sp] = pk2(f0b.z, f1b.z);
          xtw[(db + 7) * 20 + sp] = pk2(f0b.w, f1b.w);
        }
      }
      __syncthreads();

      const _Float16* ap = Ah + (size_t)(l0 + lm) * 1040 + s0 + hi * 16;
      const v8h alo = *(const v8h*)ap;
      const v8h ahi = *(const v8h*)(ap + 8);
      const v16h af = CAT16(alo, ahi);
#pragma unroll
      for (int dtt = 0; dtt < 4; ++dtt) {
        const int d = d0 + dtt * 16 + lm;
        const _Float16* bp = xT + d * 40 + hi * 16;
        const v8h blo = *(const v8h*)bp;
        const v8h bhi = *(const v8h*)(bp + 8);
        const v16h bf = CAT16(blo, bhi);
        oc[dtt] = wmma_f16(af, bf, oc[dtt]);
      }
    }

#pragma unroll
    for (int dtt = 0; dtt < 4; ++dtt) {
#pragma unroll
      for (int r = 0; r < 8; ++r) {
        const int l = l0 + r + hi * 8;
        const int d = d0 + dtt * 16 + lm;
        out_part[((size_t)b * NC + l) * DM + d] = oc[dtt][r];
      }
    }
  }
}

// ---------------------------------------------------------------------------
__global__ __launch_bounds__(256) void k_finalize(const float* __restrict__ out_part,
                                                  float* __restrict__ dst) {
  const int i = blockIdx.x * 256 + threadIdx.x;   // < 32*512
  float s = 0.f;
#pragma unroll
  for (int bb = 0; bb < BS; ++bb) s += out_part[(size_t)bb * (NC * DM) + i];
  dst[i] = s * (1.0f / 64.0f);
}

// ---------------------------------------------------------------------------
extern "C" void kernel_launch(void* const* d_in, const int* in_sizes, int n_in,
                              void* d_out, int out_size, void* d_ws, size_t ws_size,
                              hipStream_t stream) {
  (void)in_sizes; (void)n_in; (void)out_size; (void)ws_size;
  const float* x    = (const float*)d_in[0];
  const float* ce   = (const float*)d_in[1];
  const float* W    = (const float*)d_in[2];
  const float* bias = (const float*)d_in[3];
  const float* noi  = (const float*)d_in[4];
  float* out = (float*)d_out;

  char* ws = (char*)d_ws;
  float*    xemb  = (float*)ws;    ws += (size_t)BS * NV * DM * sizeof(float);
  _Float16* Wh    = (_Float16*)ws; ws += (size_t)DM * SEQ * sizeof(_Float16);
  _Float16* chv   = (_Float16*)ws; ws += (size_t)NC * DM * sizeof(_Float16);
  float*    cn    = (float*)ws;    ws += (size_t)NC * DM * sizeof(float);
  float*    mask  = (float*)ws;    ws += (size_t)NV * NC * sizeof(float);
  float*    opart = (float*)ws;    ws += (size_t)BS * NC * DM * sizeof(float);

  k_convert_w<<<(DM * SEQ) / 256, 256, 0, stream>>>(W, Wh);
  k_prep_cluster<<<NC, 256, 0, stream>>>(ce, chv, cn);
  k_embed<<<BS * 16, 512, 0, stream>>>(x, Wh, bias, xemb);
  k_prob<<<NV, 256, 0, stream>>>(xemb, cn, noi, out, mask);
  k_attn<<<BS, 512, 0, stream>>>(xemb, chv, mask, opart);
  k_finalize<<<(NC * DM) / 256, 256, 0, stream>>>(opart, out + NV * NC);
}
